// TinySSMLM_1881195676117
// MI455X (gfx1250) — matrix operand order repaired and checked by reading, not compiled, not run on hardware
//
#include <hip/hip_runtime.h>

typedef __attribute__((ext_vector_type(2)))  float        v2f;
typedef __attribute__((ext_vector_type(8)))  float        v8f;
typedef __attribute__((ext_vector_type(8)))  __bf16       v8bf;
typedef __attribute__((ext_vector_type(16))) __bf16       v16bf;
typedef __attribute__((ext_vector_type(4)))  unsigned int v4u;
typedef __attribute__((ext_vector_type(8)))  int          v8i;
typedef __attribute__((ext_vector_type(4)))  int          v4i;

// ---------------------------------------------------------------------------
// fp32 -> bf16 round-to-nearest-even
__device__ __forceinline__ unsigned short bf16_rne(float x) {
    unsigned int u = __float_as_uint(x);
    u += 0x7FFFu + ((u >> 16) & 1u);
    return (unsigned short)(u >> 16);
}
__device__ __forceinline__ float bf16_to_f32(unsigned short h) {
    return __uint_as_float(((unsigned int)h) << 16);
}

// ---------------------------------------------------------------------------
// TDM: DMA a 2D tile (tile_d1 rows x tile_d0 elems, 4-byte elements) from a
// row-major [tensor_d1, tensor_d0] fp32 tensor into LDS (packed, no pad).
// D# bitfields per CDNA5 ISA 8.3-8.6; trailing groups zero (2-D tensor).
// Wave-level op tracked by TENSORcnt. 6-arg builtin (amdgpu-toolchain form).
__device__ __forceinline__ void tdm_load_tile_2d_f32(
    unsigned lds_off, const void* gaddr,
    unsigned tensor_d0, unsigned tensor_d1,
    unsigned tile_d0, unsigned tile_d1, unsigned stride_d0)
{
    unsigned long long ga = (unsigned long long)(size_t)gaddr;
    v4u g0;
    g0[0] = 1u;                                         // count=1 (valid user D#)
    g0[1] = lds_off;                                    // lds_addr [63:32]
    g0[2] = (unsigned)(ga & 0xFFFFFFFFu);               // global_addr[31:0]
    g0[3] = (unsigned)((ga >> 32) & 0x1FFFFFFu)         // global_addr[56:32]
          | (2u << 30);                                 // type=2 ("image")
    v8i g1;
    g1[0] = (int)(2u << 16);                            // data_size=4B, no mask
    g1[1] = (int)((tensor_d0 & 0xFFFFu) << 16);         // tensor_dim0[15:0]
    g1[2] = (int)(((tensor_d0 >> 16) & 0xFFFFu)         // tensor_dim0[31:16]
          |       ((tensor_d1 & 0xFFFFu) << 16));       // tensor_dim1[15:0]
    g1[3] = (int)(((tensor_d1 >> 16) & 0xFFFFu)         // tensor_dim1[31:16]
          |       (tile_d0 << 16));                     // tile_dim0
    g1[4] = (int)(tile_d1 & 0xFFFFu);                   // tile_dim1 (tile_dim2=0)
    g1[5] = (int)stride_d0;                             // tensor_dim0_stride[31:0]
    g1[6] = 0;                                          // stride0 hi / stride1 lo
    g1[7] = 0;
    v4i z4 = {0, 0, 0, 0};
    v8i z8 = {0, 0, 0, 0, 0, 0, 0, 0};
    __builtin_amdgcn_tensor_load_to_lds(g0, g1, z4, z4, z8, 0);
}

// ===========================================================================
// Small-GEMM path (exact fp32): C[M,N] = A1@B1 (+A2@B2) (+bias[N])
// V_WMMA_F32_16X16X4_F32, 64x32 block tile, KT=32, 8 waves.
// B tile staged by the Tensor Data Mover (async DMA -> LDS, TENSORcnt);
// A tile staged by threads (needs odd-pitch pad for conflict-free col reads).
// ===========================================================================
#define MT 64
#define NT 32
#define KT 32

__global__ __launch_bounds__(256) void gemm_wmma_f32(
    const float* __restrict__ A1, const float* __restrict__ B1,
    const float* __restrict__ A2, const float* __restrict__ B2,
    const float* __restrict__ bias, float* __restrict__ C,
    int M, int N, int K)
{
    __shared__ float As[MT][KT + 1];
    __shared__ float Bs[KT][NT];     // packed: filled by TDM, read row-wise

    const int tid  = threadIdx.x;
    const int wave = tid >> 5;
    const int lane = tid & 31;
    const int m0 = blockIdx.y * MT;
    const int n0 = blockIdx.x * NT;
    const int wm = (wave >> 1) * 16;
    const int wn = (wave & 1) * 16;
    const int frow = lane & 15;
    const int kb   = (lane >> 4) << 1;
    const unsigned bs_lds = (unsigned)(size_t)&Bs[0][0];  // LDS aperture: addr[31:0]

    v8f acc = {0.f, 0.f, 0.f, 0.f, 0.f, 0.f, 0.f, 0.f};

    for (int pass = 0; pass < 2; ++pass) {
        const float* __restrict__ A = pass ? A2 : A1;
        const float* __restrict__ B = pass ? B2 : B1;
        if (A == nullptr) break;

        for (int k0 = 0; k0 < K; k0 += KT) {
            __syncthreads();   // protect LDS (incl. Bs) from WAR vs prev chunk
            // ---- B tile via Tensor Data Mover (wave 0 issues the DMA) ----
            if (tid < 32) {
                tdm_load_tile_2d_f32(bs_lds, &B[(size_t)k0 * N + n0],
                                     (unsigned)N, (unsigned)K,
                                     NT, KT, (unsigned)N);
            }
            // ---- A tile staged by all 8 waves (overlaps with the DMA) ----
            #pragma unroll
            for (int i = 0; i < (MT * KT) / 256; ++i) {
                int idx = tid + i * 256;
                int r = idx >> 5, c = idx & (KT - 1);
                As[r][c] = A[(size_t)(m0 + r) * K + (k0 + c)];
            }
            if (tid < 32) __builtin_amdgcn_s_wait_tensorcnt(0);
            __syncthreads();

            #pragma unroll
            for (int kk = 0; kk < KT; kk += 4) {
                v2f a, b;
                a.x = As[wm + frow][kk + kb];
                a.y = As[wm + frow][kk + kb + 1];
                b.x = Bs[kk + kb][wn + frow];
                b.y = Bs[kk + kb + 1][wn + frow];
                acc = __builtin_amdgcn_wmma_f32_16x16x4_f32(
                        false, a, false, b, (short)0, acc, false, false);
            }
        }
    }

    const int cn = n0 + wn + frow;
    float bv = (bias != nullptr) ? bias[cn] : 0.f;
    const int mhalf = (lane >> 4) * 8;
    #pragma unroll
    for (int r = 0; r < 8; ++r) {
        C[(size_t)(m0 + wm + mhalf + r) * N + cn] = acc[r] + bv;
    }
}

// ===========================================================================
// Head-GEMM path: bf16 split-precision (2-term Ozaki) fp32-accurate GEMM.
// C = A@B + bias; on-the-fly hi/lo bf16 split into LDS;
// V_WMMA_F32_16X16X32_BF16, 3 products per tile (hh + hl + lh).
// Block tile 128x64 (8 waves, 4x2, wave tile 32x32), KT = 32.
// ===========================================================================
#define HMT 128
#define HNT 64
#define HKT 32
#define APITCH 40
#define BPITCH 40

__device__ __forceinline__ v16bf frag_from_runs(const unsigned short* p0,
                                                const unsigned short* p1) {
    v8bf x = *(const v8bf*)p0;
    v8bf y = *(const v8bf*)p1;
    return __builtin_shufflevector(x, y, 0, 1, 2, 3, 4, 5, 6, 7,
                                         8, 9, 10, 11, 12, 13, 14, 15);
}

__global__ __launch_bounds__(256) void gemm_wmma_bf16split(
    const float* __restrict__ A, const float* __restrict__ Bm,
    const float* __restrict__ bias, float* __restrict__ C,
    int M, int N, int K)
{
    __shared__ unsigned short AsHi[HMT * APITCH];
    __shared__ unsigned short AsLo[HMT * APITCH];
    __shared__ unsigned short BtHi[HNT * BPITCH];
    __shared__ unsigned short BtLo[HNT * BPITCH];
    unsigned int* AsHi32 = (unsigned int*)AsHi;
    unsigned int* AsLo32 = (unsigned int*)AsLo;

    const int tid  = threadIdx.x;
    const int wave = tid >> 5;
    const int lane = tid & 31;
    const int m0 = blockIdx.y * HMT;
    const int n0 = blockIdx.x * HNT;
    const int wm = (wave >> 1) * 32;
    const int wn = (wave & 1) * 32;
    const int frow = lane & 15;
    const int hl   = lane >> 4;

    v8f acc00 = {0.f,0.f,0.f,0.f,0.f,0.f,0.f,0.f};
    v8f acc01 = acc00, acc10 = acc00, acc11 = acc00;

    for (int k0 = 0; k0 < K; k0 += HKT) {
        __syncthreads();
        // ---- stage A 128x32: split hi/lo bf16, packed-pair u32 stores ----
        #pragma unroll
        for (int i = 0; i < 8; ++i) {
            int idx = tid + i * 256;
            int r   = idx >> 4;
            int c2  = (idx & 15) * 2;
            float2 x = *(const float2*)&A[(size_t)(m0 + r) * K + k0 + c2];
            unsigned short h0 = bf16_rne(x.x), h1 = bf16_rne(x.y);
            unsigned short l0 = bf16_rne(x.x - bf16_to_f32(h0));
            unsigned short l1 = bf16_rne(x.y - bf16_to_f32(h1));
            int o = r * (APITCH / 2) + (c2 >> 1);
            AsHi32[o] = (unsigned int)h0 | ((unsigned int)h1 << 16);
            AsLo32[o] = (unsigned int)l0 | ((unsigned int)l1 << 16);
        }
        // ---- stage B 32x64 transposed, split hi/lo ----
        #pragma unroll
        for (int i = 0; i < 4; ++i) {
            int idx = tid + i * 256;
            int r   = idx >> 5;
            int c2  = (idx & 31) * 2;
            float2 x = *(const float2*)&Bm[(size_t)(k0 + r) * N + n0 + c2];
            unsigned short h0 = bf16_rne(x.x), h1 = bf16_rne(x.y);
            unsigned short l0 = bf16_rne(x.x - bf16_to_f32(h0));
            unsigned short l1 = bf16_rne(x.y - bf16_to_f32(h1));
            BtHi[(c2    ) * BPITCH + r] = h0;
            BtHi[(c2 + 1) * BPITCH + r] = h1;
            BtLo[(c2    ) * BPITCH + r] = l0;
            BtLo[(c2 + 1) * BPITCH + r] = l1;
        }
        if (k0 + HKT < K) {
            __builtin_prefetch(&A[(size_t)(m0 + (tid >> 4)) * K + k0 + HKT + (tid & 15) * 2], 0, 1);
            __builtin_prefetch(&Bm[(size_t)(k0 + HKT + (tid >> 5)) * N + n0 + (tid & 31) * 2], 0, 1);
        }
        __syncthreads();

        // A frag: half j -> K = 16*(j>=8) + 8*(lane>=16) + (j&7)
        const unsigned short* ar0  = &AsHi[(wm + frow) * APITCH];
        const unsigned short* ar1  = &AsHi[(wm + 16 + frow) * APITCH];
        const unsigned short* al0p = &AsLo[(wm + frow) * APITCH];
        const unsigned short* al1p = &AsLo[(wm + 16 + frow) * APITCH];
        v16bf ah0 = frag_from_runs(ar0 + 8 * hl, ar0 + 16 + 8 * hl);
        v16bf ah1 = frag_from_runs(ar1 + 8 * hl, ar1 + 16 + 8 * hl);
        v16bf al0 = frag_from_runs(al0p + 8 * hl, al0p + 16 + 8 * hl);
        v16bf al1 = frag_from_runs(al1p + 8 * hl, al1p + 16 + 8 * hl);
        // B frag: half j -> K = 16*(j>=8) + 8*(lane>=16) + (j&7), N = lane&15
        const unsigned short* br0  = &BtHi[(wn + frow) * BPITCH];
        const unsigned short* br1  = &BtHi[(wn + 16 + frow) * BPITCH];
        const unsigned short* bl0p = &BtLo[(wn + frow) * BPITCH];
        const unsigned short* bl1p = &BtLo[(wn + 16 + frow) * BPITCH];
        v16bf bh0 = frag_from_runs(br0 + 8 * hl, br0 + 16 + 8 * hl);
        v16bf bh1 = frag_from_runs(br1 + 8 * hl, br1 + 16 + 8 * hl);
        v16bf bl0 = frag_from_runs(bl0p + 8 * hl, bl0p + 16 + 8 * hl);
        v16bf bl1 = frag_from_runs(bl1p + 8 * hl, bl1p + 16 + 8 * hl);

        acc00 = __builtin_amdgcn_wmma_f32_16x16x32_bf16(false, ah0, false, bh0, (short)0, acc00, false, false);
        acc00 = __builtin_amdgcn_wmma_f32_16x16x32_bf16(false, ah0, false, bl0, (short)0, acc00, false, false);
        acc00 = __builtin_amdgcn_wmma_f32_16x16x32_bf16(false, al0, false, bh0, (short)0, acc00, false, false);

        acc01 = __builtin_amdgcn_wmma_f32_16x16x32_bf16(false, ah0, false, bh1, (short)0, acc01, false, false);
        acc01 = __builtin_amdgcn_wmma_f32_16x16x32_bf16(false, ah0, false, bl1, (short)0, acc01, false, false);
        acc01 = __builtin_amdgcn_wmma_f32_16x16x32_bf16(false, al0, false, bh1, (short)0, acc01, false, false);

        acc10 = __builtin_amdgcn_wmma_f32_16x16x32_bf16(false, ah1, false, bh0, (short)0, acc10, false, false);
        acc10 = __builtin_amdgcn_wmma_f32_16x16x32_bf16(false, ah1, false, bl0, (short)0, acc10, false, false);
        acc10 = __builtin_amdgcn_wmma_f32_16x16x32_bf16(false, al1, false, bh0, (short)0, acc10, false, false);

        acc11 = __builtin_amdgcn_wmma_f32_16x16x32_bf16(false, ah1, false, bh1, (short)0, acc11, false, false);
        acc11 = __builtin_amdgcn_wmma_f32_16x16x32_bf16(false, ah1, false, bl1, (short)0, acc11, false, false);
        acc11 = __builtin_amdgcn_wmma_f32_16x16x32_bf16(false, al1, false, bh1, (short)0, acc11, false, false);
    }

    const int mhalf = hl * 8;
    #pragma unroll
    for (int ni = 0; ni < 2; ++ni) {
        const int cn = n0 + wn + ni * 16 + frow;
        const float bv = bias[cn];
        #pragma unroll
        for (int mi = 0; mi < 2; ++mi) {
            const v8f& a = (ni == 0) ? (mi == 0 ? acc00 : acc10)
                                     : (mi == 0 ? acc01 : acc11);
            const int cmb = m0 + wm + mi * 16 + mhalf;
            #pragma unroll
            for (int r = 0; r < 8; ++r) {
                __builtin_nontemporal_store(a[r] + bv,
                    &C[(size_t)(cmb + r) * N + cn]);
            }
        }
    }
}

// ===========================================================================
// embed gather + double RMSNorm (one block per token)
// ===========================================================================
__global__ __launch_bounds__(256) void embed_rmsnorm2(
    const int* __restrict__ ids, const float* __restrict__ emb,
    const float* __restrict__ w1, const float* __restrict__ w2,
    float* __restrict__ u, int D)
{
    __shared__ float red[256];
    const int row = blockIdx.x;
    const int tid = threadIdx.x;
    const int per = D >> 8;
    const float* e = emb + (size_t)ids[row] * D;

    float v[8];
    float ss = 0.f;
    for (int i = 0; i < per; ++i) {
        v[i] = e[tid + i * 256];
        ss += v[i] * v[i];
    }
    red[tid] = ss; __syncthreads();
    for (int off = 128; off > 0; off >>= 1) {
        if (tid < off) red[tid] += red[tid + off];
        __syncthreads();
    }
    const float r1 = rsqrtf(red[0] / (float)D + 1e-6f);
    __syncthreads();

    float ss2 = 0.f;
    for (int i = 0; i < per; ++i) {
        v[i] = v[i] * r1 * w1[tid + i * 256];
        ss2 += v[i] * v[i];
    }
    red[tid] = ss2; __syncthreads();
    for (int off = 128; off > 0; off >>= 1) {
        if (tid < off) red[tid] += red[tid + off];
        __syncthreads();
    }
    const float r2 = rsqrtf(red[0] / (float)D + 1e-6f);

    for (int i = 0; i < per; ++i)
        u[(size_t)row * D + tid + i * 256] = v[i] * r2 * w2[tid + i * 256];
}

// ===========================================================================
// complex linear scan: xs[l] = A*xs[l-1] + uB[l]   (== causal conv with A^t)
// ===========================================================================
__global__ void ssm_scan(
    const float* __restrict__ uB_re, const float* __restrict__ uB_im,
    const float* __restrict__ A_re,  const float* __restrict__ A_im,
    float* __restrict__ xs_re, float* __restrict__ xs_imn,
    int L, int S)
{
    const int b = blockIdx.x;
    const int s = threadIdx.x;
    const float ar = A_re[s], ai = A_im[s];
    const size_t base = (size_t)b * L * S + s;

    float cr = 0.f, ci = 0.f;
    for (int l = 0; l < L; ++l) {
        const size_t off = base + (size_t)l * S;
        const float xr = uB_re[off];
        const float xi = uB_im[off];
        const float nr = ar * cr - ai * ci + xr;
        const float ni = ar * ci + ai * cr + xi;
        cr = nr; ci = ni;
        xs_re[off]  = cr;
        xs_imn[off] = -ci;
    }
}

// ===========================================================================
extern "C" void kernel_launch(void* const* d_in, const int* in_sizes, int n_in,
                              void* d_out, int out_size, void* d_ws, size_t ws_size,
                              hipStream_t stream)
{
    const int*   ids   = (const int*)  d_in[0];
    const float* emb   = (const float*)d_in[1];
    const float* w1    = (const float*)d_in[2];
    const float* w2    = (const float*)d_in[3];
    const float* A_re  = (const float*)d_in[4];
    const float* A_im  = (const float*)d_in[5];
    const float* B_re  = (const float*)d_in[6];
    const float* B_im  = (const float*)d_in[7];
    const float* C_re  = (const float*)d_in[8];
    const float* C_im  = (const float*)d_in[9];
    const float* D_p   = (const float*)d_in[10];
    const float* headw = (const float*)d_in[11];
    const float* headb = (const float*)d_in[12];
    float* out = (float*)d_out;

    const int BL = in_sizes[0];        // 4096
    const int D  = in_sizes[2];        // 1024
    const int S  = in_sizes[4];        // 256
    const int V  = in_sizes[12];       // 32000
    const int Bz = 2;                  // reference batch size
    const int L  = BL / Bz;            // 2048

    float* ws    = (float*)d_ws;
    float* u     = ws;  ws += (size_t)BL * D;
    float* uBre  = ws;  ws += (size_t)BL * S;
    float* uBim  = ws;  ws += (size_t)BL * S;
    float* xsre  = ws;  ws += (size_t)BL * S;
    float* xsimn = ws;  ws += (size_t)BL * S;
    float* y     = ws;  ws += (size_t)BL * D;

    // 1) embed + RMSNorm(embed_w) + RMSNorm(ssm_w)
    embed_rmsnorm2<<<BL, 256, 0, stream>>>(ids, emb, w1, w2, u, D);

    // 2) uB = u @ B (complex: two exact-f32 WMMA GEMMs, B tiles via TDM)
    gemm_wmma_f32<<<dim3(S / NT, BL / MT), 256, 0, stream>>>(
        u, B_re, nullptr, nullptr, nullptr, uBre, BL, S, D);
    gemm_wmma_f32<<<dim3(S / NT, BL / MT), 256, 0, stream>>>(
        u, B_im, nullptr, nullptr, nullptr, uBim, BL, S, D);

    // 3) causal conv with A^t == linear recurrence scan
    ssm_scan<<<Bz, S, 0, stream>>>(uBre, uBim, A_re, A_im, xsre, xsimn, L, S);

    // 4) y = xs_re@C_re + (-xs_im)@C_im + D_p (fused dual GEMM, exact f32)
    gemm_wmma_f32<<<dim3(D / NT, BL / MT), 256, 0, stream>>>(
        xsre, C_re, xsimn, C_im, D_p, y, BL, D, S);

    // 5) logits = y @ head_w + head_b : dominant 268 GFLOP on the bf16 matrix
    //    pipe (split-precision), nontemporal logits stores keep the 131MB
    //    head_w L2-resident across the 32 row-blocks.
    gemm_wmma_bf16split<<<dim3(V / HNT, BL / HMT), 256, 0, stream>>>(
        y, headw, headb, out, BL, V, D);
}
